// GaussianBiRNN_11012296147396
// MI455X (gfx1250) — compile-verified, matrix operand order checked by reading, not run on hardware
//
#include <hip/hip_runtime.h>
#include <cstdint>

// ---------------------------------------------------------------------------
// Types for CDNA5 WMMA (wave32): v_wmma_f32_16x16x32_bf16
// ---------------------------------------------------------------------------
typedef __bf16 bf16;
typedef bf16  v16bf __attribute__((ext_vector_type(16)));
typedef float v8f   __attribute__((ext_vector_type(8)));

// TDM descriptor vector types (per CDNA5_HIP.md probe results)
typedef unsigned int u32x4 __attribute__((ext_vector_type(4)));
typedef int          i32x4 __attribute__((ext_vector_type(4)));
typedef int          i32x8 __attribute__((ext_vector_type(8)));

// Load a 16x32 bf16 fragment (A or B^T) from row-major [rows, ld] storage.
// ISA layout (16-bit A matrix 16x32): lanes 0-15 hold K={k..k+7, k+16..k+23}
// for row M=lane; lanes 16-31 hold K={k+8..k+15, k+24..k+31} for row M=lane-16.
__device__ inline v16bf load_frag(const bf16* __restrict__ base, int ld,
                                  int row0, int k0, int lane) {
    int r    = row0 + (lane & 15);
    int ksub = (lane >> 4) << 3;               // 0 or 8
    const bf16* p = base + (long)r * ld + k0 + ksub;
    union { v16bf v; uint4 q[2]; } u;
    u.q[0] = *(const uint4*)(p);               // 8 bf16 = 16B
    u.q[1] = *(const uint4*)(p + 16);          // next K-chunk (+16 elements)
    return u.v;
}

__device__ inline v8f wmma_bf16(v16bf a, v16bf b, v8f c) {
    return __builtin_amdgcn_wmma_f32_16x16x32_bf16(
        /*neg_a=*/false, a, /*neg_b=*/false, b,
        /*c_mod=*/(short)0, c, /*reuse_a=*/false, /*reuse_b=*/false);
}

// ---------------------------------------------------------------------------
// Tensor Data Mover: 2D tile load (global -> LDS), bf16 elements.
//   width_elems : contiguous elements per row (dim0)
//   rows        : number of rows (dim1)
//   stride_elems: row stride in the global tensor (dim0_stride)
// Builds D# group0/group1 per CDNA5 ISA.
// ---------------------------------------------------------------------------
#if __has_builtin(__builtin_amdgcn_tensor_load_to_lds)
#define HAVE_TDM 1
#endif

#ifdef HAVE_TDM
__device__ inline void tdm_load_tile_2d(unsigned lds_off_bytes, const void* gaddr,
                                        unsigned width_elems, unsigned rows,
                                        unsigned stride_elems) {
    unsigned long ga = (unsigned long)gaddr;
    u32x4 g0;
    g0[0] = 1u;                                       // count=1 (valid), no gather
    g0[1] = lds_off_bytes;                            // lds_addr [63:32]
    g0[2] = (unsigned)(ga & 0xffffffffu);             // global_addr[31:0]
    g0[3] = (unsigned)((ga >> 32) & 0x01ffffffu)      // global_addr[56:32]
          | (2u << 30);                               // type=2 ("image") [127:126]
    i32x8 g1;
    g1[0] = (int)(1u << 16);                          // wg_mask=0, data_size=1 (2B)
    g1[1] = (int)((width_elems & 0xffffu) << 16);     // tensor_dim0[15:0]  -> [63:48]
    g1[2] = (int)((width_elems >> 16) & 0xffffu)      // tensor_dim0[31:16] -> [79:64]
          | (int)((rows & 0xffffu) << 16);            // tensor_dim1[15:0]  -> [95:80]
    g1[3] = (int)((width_elems & 0xffffu) << 16);     // tile_dim0 -> [127:112]
    g1[4] = (int)(rows & 0xffffu);                    // tile_dim1 -> [143:128]; tile_dim2=0
    g1[5] = (int)stride_elems;                        // tensor_dim0_stride[31:0] -> [191:160]
    g1[6] = 0;                                        // stride[47:32], dim1_stride lo
    g1[7] = 0;
    i32x4 z4 = {0, 0, 0, 0};
#if defined(__clang_major__) && (__clang_major__ >= 23)
    i32x8 z8 = {0, 0, 0, 0, 0, 0, 0, 0};
    __builtin_amdgcn_tensor_load_to_lds(g0, g1, z4, z4, z8, 0);   // 6-arg form
#else
    __builtin_amdgcn_tensor_load_to_lds(g0, g1, z4, z4, 0);       // 5-arg form
#endif
}
#endif

// ---------------------------------------------------------------------------
// Utility kernels
// ---------------------------------------------------------------------------
__global__ void cvt_f32_to_bf16_kernel(const float* __restrict__ src,
                                       bf16* __restrict__ dst, long n) {
    long i = (long)blockIdx.x * blockDim.x + threadIdx.x;
    if (i < n) dst[i] = (bf16)src[i];
}

// W[K,N] f32  ->  Wt[N,K] bf16   (one-time; weights are tiny vs. activations)
__global__ void transpose_cvt_kernel(const float* __restrict__ W,
                                     bf16* __restrict__ Wt, int K, int N) {
    long i = (long)blockIdx.x * blockDim.x + threadIdx.x;
    long total = (long)K * N;
    if (i >= total) return;
    int n = (int)(i / K);
    int k = (int)(i % K);
    Wt[i] = (bf16)W[(long)k * N + n];
}

__global__ void zero_ctr_kernel(unsigned* c) {
    if (threadIdx.x < 2) c[threadIdx.x] = 0u;
}

// ---------------------------------------------------------------------------
// Generic bf16 GEMM:  C[M,N] = act(A[M,K] @ B[K,N] + bias)
//   A   : bf16 row-major [M,K]
//   Bt  : bf16 row-major [N,K]  (B pre-transposed)
//   Cf  : optional f32 output, Cb : optional bf16 output
// Block = 256 threads = 8 waves arranged 2(M) x 4(N); wave tile = 16x16.
// All problem dims here are multiples of 16.
// ---------------------------------------------------------------------------
__global__ void __launch_bounds__(256)
gemm_bf16_kernel(const bf16* __restrict__ A, const bf16* __restrict__ Bt,
                 const float* __restrict__ bias,
                 float* __restrict__ Cf, bf16* __restrict__ Cb,
                 int M, int N, int K, int act) {
    int wave = threadIdx.x >> 5;
    int lane = threadIdx.x & 31;
    int row0 = (blockIdx.y * 2 + (wave >> 2)) * 16;
    int col0 = (blockIdx.x * 4 + (wave & 3)) * 16;
    if (row0 >= M || col0 >= N) return;

    v8f acc = {};
    for (int k = 0; k < K; k += 32) {
        // speculative prefetch of the next K-chunk (global_prefetch_b8)
        __builtin_prefetch(A + (long)(row0 + (lane & 15)) * K + k + 32, 0, 3);
        __builtin_prefetch(Bt + (long)(col0 + (lane & 15)) * K + k + 32, 0, 3);
        v16bf af = load_frag(A,  K, row0, k, lane);
        v16bf bf = load_frag(Bt, K, col0, k, lane);
        acc = wmma_bf16(af, bf, acc);
    }

    int col = col0 + (lane & 15);
    float bv = bias ? bias[col] : 0.0f;
#pragma unroll
    for (int j = 0; j < 8; ++j) {
        int row = row0 + j + ((lane >> 4) << 3);
        float v = acc[j] + bv;
        if (act) v = fmaxf(v, 0.0f);
        if (Cf) Cf[(long)row * N + col] = v;
        if (Cb) Cb[(long)row * N + col] = (bf16)v;
    }
}

// ---------------------------------------------------------------------------
// Persistent bidirectional GRU recurrence.
//   16 workgroups: blocks [0..7] = forward dir, [8..15] = backward dir.
//   WG s owns output columns [s*64, s*64+64) of h; holds the matching
//   512x192 bf16 slice of Wh^T (r|z|n column groups) in LDS (192 KB,
//   loaded via the Tensor Data Mover). Per step: slice-GEMM
//   [32,512]x[512,192] via WMMA (B frags from LDS), stage G into LDS,
//   elementwise GRU update, double-buffered h in global, atomic sub-grid
//   barrier per direction per step.
// ---------------------------------------------------------------------------
#define GRU_S   8
#define GRU_T   512
#define GRU_B   32
#define GRU_H   512
#define GRU_SMEM (192 * 512 * 2 + 32 * 192 * 4)   // Wh slice + G staging

__device__ inline void subgrid_barrier(unsigned* ctr, unsigned nwg, unsigned phase) {
    __syncthreads();
    if (threadIdx.x == 0) {
        __threadfence();
        atomicAdd(ctr, 1u);
        unsigned target = nwg * (phase + 1u);
        while (atomicAdd(ctr, 0u) < target) { __builtin_amdgcn_s_sleep(1); }
        __threadfence();
    }
    __syncthreads();
}

__global__ void __launch_bounds__(256)
gru_kernel(const float* __restrict__ xpf, const float* __restrict__ xpb,
           const float* __restrict__ fbhn, const float* __restrict__ bbhn,
           const bf16* __restrict__ fWhT, const bf16* __restrict__ bWhT,
           float* __restrict__ hf,      // [dir][parity][32][512] f32
           bf16* __restrict__ hb,       // [dir][parity][32][512] bf16
           bf16* __restrict__ hcat,     // [BT][1024] bf16
           unsigned* __restrict__ counters) {
    extern __shared__ char smem_raw[];
    bf16*  sW = (bf16*)smem_raw;                       // [192][512] bf16
    float* sG = (float*)(smem_raw + 192 * 512 * 2);    // [32][192]  f32

    const int tid  = threadIdx.x;
    const int dir  = (blockIdx.x >= GRU_S) ? 1 : 0;
    const int s    = blockIdx.x - dir * GRU_S;
    const float* xp  = dir ? xpb : xpf;
    const float* bhn = dir ? bbhn : fbhn;
    const bf16*  WhT = dir ? bWhT : fWhT;
    unsigned* ctr = counters + dir;

    // Preload Wh^T slice into LDS. LDS row n (group g = n/64, j = n%64)
    // <- global row g*512 + s*64 + j (g: 0=r, 1=z, 2=n column groups).
    // Each group is a contiguous 64-row x 512-col 2D tile -> one TDM DMA.
#ifdef HAVE_TDM
    if ((tid >> 5) == 0) {                 // single wave issues the 3 DMAs
#pragma unroll
        for (int g = 0; g < 3; ++g) {
            const bf16* src = WhT + ((long)g * 512 + s * 64) * 512;
            tdm_load_tile_2d((unsigned)(g * 64 * 512 * 2), src,
                             /*width=*/512, /*rows=*/64, /*stride=*/512);
        }
        __builtin_amdgcn_s_wait_tensorcnt(0);
    }
#else
    for (int i = tid; i < 192 * 64; i += 256) {        // 64 x uint4 per row
        int n = i >> 6, c = i & 63;
        int g = n >> 6, j = n & 63;
        long grow = (long)g * 512 + s * 64 + j;
        *(uint4*)(sW + n * 512 + c * 8) =
            *(const uint4*)(WhT + grow * 512 + c * 8);
    }
#endif

    // Zero this WG's slice of h (read-parity-0 buffers).
    {
        float* h0  = hf + (long)(dir * 2 + 0) * GRU_B * GRU_H;
        bf16*  h0b = hb + (long)(dir * 2 + 0) * GRU_B * GRU_H;
        for (int i = tid; i < GRU_B * 64; i += 256) {
            int bi = i >> 6, j = i & 63;
            int col = s * 64 + j;
            h0[bi * GRU_H + col]  = 0.0f;
            h0b[bi * GRU_H + col] = (bf16)0.0f;
        }
    }
    __threadfence();
    subgrid_barrier(ctr, GRU_S, 0);        // publishes LDS (TDM) + h0 to all

    const int wave = tid >> 5, lane = tid & 31;

    for (int t = 0; t < GRU_T; ++t) {
        const int tt = dir ? (GRU_T - 1 - t) : t;
        const int pr = t & 1;                      // read parity
        const bf16*  hrdb = hb + (long)(dir * 2 + pr) * GRU_B * GRU_H;
        const float* hrd  = hf + (long)(dir * 2 + pr) * GRU_B * GRU_H;
        bf16*  hwrb = hb + (long)(dir * 2 + (pr ^ 1)) * GRU_B * GRU_H;
        float* hwr  = hf + (long)(dir * 2 + (pr ^ 1)) * GRU_B * GRU_H;

        // ---- G = h @ Wh_slice : M=32, K=512, N=192 (2 x 12 tiles, 8 waves)
        for (int ti = wave; ti < 24; ti += 8) {
            int mt = ti / 12, nt = ti % 12;
            v8f acc = {};
            for (int k = 0; k < 512; k += 32) {
                v16bf af = load_frag(hrdb, GRU_H, mt * 16, k, lane); // global
                v16bf bf = load_frag(sW, 512, nt * 16, k, lane);     // LDS
                acc = wmma_bf16(af, bf, acc);
            }
            int col = nt * 16 + (lane & 15);
#pragma unroll
            for (int j = 0; j < 8; ++j) {
                int row = mt * 16 + j + ((lane >> 4) << 3);
                sG[row * 192 + col] = acc[j];
            }
        }
        __syncthreads();

        // ---- elementwise GRU update for the owned 32x64 slice
        for (int i = tid; i < GRU_B * 64; i += 256) {
            int bi = i >> 6, j = i & 63;
            int col = s * 64 + j;
            long token = (long)bi * GRU_T + tt;
            const float* xr = xp + token * 1536;
            float gr = sG[bi * 192 + j];
            float gz = sG[bi * 192 + 64 + j];
            float gn = sG[bi * 192 + 128 + j];
            float r = 1.0f / (1.0f + __expf(-(xr[col] + gr)));
            float z = 1.0f / (1.0f + __expf(-(xr[512 + col] + gz)));
            float n = tanhf(xr[1024 + col] + r * (gn + bhn[col]));
            float hold = hrd[bi * GRU_H + col];
            float hnew = (1.0f - z) * n + z * hold;
            hwr[bi * GRU_H + col]  = hnew;
            hwrb[bi * GRU_H + col] = (bf16)hnew;
            hcat[token * 1024 + dir * 512 + col] = (bf16)hnew;
        }
        __threadfence();
        subgrid_barrier(ctr, GRU_S, t + 1);
    }
}

// ---------------------------------------------------------------------------
// Q = L @ L^T per token from vflat[528] (32 diag params + 496 strict-lower).
// One wave per token; L rows staged in LDS.
// ---------------------------------------------------------------------------
__global__ void __launch_bounds__(256)
qbuild_kernel(const float* __restrict__ vflat, float* __restrict__ Qout) {
    __shared__ float sL[8 * 32 * 32];
    int wave = threadIdx.x >> 5;
    int lane = threadIdx.x & 31;
    long token = (long)blockIdx.x * 8 + wave;
    const float* vr = vflat + token * 528;
    float* Lw = sL + wave * 1024;

    int i = lane;  // row of L owned by this lane
    for (int c = 0; c < 32; ++c) {
        float v;
        if (c < i)       v = vr[32 + (i * (i - 1)) / 2 + c];
        else if (c == i) v = __expf(vr[i]) + 1e-4f;
        else             v = 0.0f;
        Lw[i * 32 + c] = v;
    }
    __syncthreads();

    float* qr = Qout + token * 1024 + i * 32;
    for (int j = 0; j < 32; ++j) {
        float acc = 0.0f;
        for (int k = 0; k <= j; ++k)          // L[i][k]==0 for k>i anyway
            acc += Lw[i * 32 + k] * Lw[j * 32 + k];
        qr[j] = acc;
    }
}

// ---------------------------------------------------------------------------
// Host-side orchestration
// ---------------------------------------------------------------------------
extern "C" void kernel_launch(void* const* d_in, const int* in_sizes, int n_in,
                              void* d_out, int out_size, void* d_ws, size_t ws_size,
                              hipStream_t stream) {
    const int Bsz = 32, T = 512, BT = Bsz * T, Din = 128, H = 512, Hm = 256;
    const int d = 32, NV = 528, H3 = 1536;

    const float* x    = (const float*)d_in[0];
    const float* iW1  = (const float*)d_in[1];
    const float* ib1  = (const float*)d_in[2];
    const float* iW2  = (const float*)d_in[3];
    const float* ib2  = (const float*)d_in[4];
    const float* fWi  = (const float*)d_in[5];
    const float* fbi  = (const float*)d_in[6];
    const float* fWh  = (const float*)d_in[7];
    const float* fbhn = (const float*)d_in[8];
    const float* bWi  = (const float*)d_in[9];
    const float* bbi  = (const float*)d_in[10];
    const float* bWh  = (const float*)d_in[11];
    const float* bbhn = (const float*)d_in[12];
    const float* mW1  = (const float*)d_in[13];
    const float* mb1  = (const float*)d_in[14];
    const float* mW2  = (const float*)d_in[15];
    const float* mb2  = (const float*)d_in[16];
    const float* vW1  = (const float*)d_in[17];
    const float* vb1  = (const float*)d_in[18];
    const float* vW2  = (const float*)d_in[19];
    const float* vb2  = (const float*)d_in[20];
    const float* dW1  = (const float*)d_in[21];
    const float* db1  = (const float*)d_in[22];
    const float* dW2  = (const float*)d_in[23];
    const float* db2  = (const float*)d_in[24];

    float* outA = (float*)d_out;                    // [BT,1024]
    float* outB = outA + (long)BT * d * d;          // [BT,32]
    float* outQ = outB + (long)BT * d;              // [BT,1024]

    // ---- workspace layout (256B aligned) ----
    char* ws = (char*)d_ws;
    size_t off = 0;
    auto alloc = [&](size_t bytes) -> char* {
        char* p = ws + off;
        off = (off + bytes + 255) & ~(size_t)255;
        return p;
    };
    bf16* xb    = (bf16*)alloc((size_t)BT * Din * 2);
    bf16* t1b   = (bf16*)alloc((size_t)BT * Hm * 2);
    bf16* ub    = (bf16*)alloc((size_t)BT * H * 2);
    float* xpf  = (float*)alloc((size_t)BT * H3 * 4);
    float* xpb  = (float*)alloc((size_t)BT * H3 * 4);
    bf16* iW1T  = (bf16*)alloc((size_t)Hm * Din * 2);
    bf16* iW2T  = (bf16*)alloc((size_t)H * Hm * 2);
    bf16* fWiT  = (bf16*)alloc((size_t)H3 * H * 2);
    bf16* fWhT  = (bf16*)alloc((size_t)H3 * H * 2);
    bf16* bWiT  = (bf16*)alloc((size_t)H3 * H * 2);
    bf16* bWhT  = (bf16*)alloc((size_t)H3 * H * 2);
    bf16* mW1T  = (bf16*)alloc((size_t)Hm * (2 * H) * 2);
    bf16* mW2T  = (bf16*)alloc((size_t)d * Hm * 2);
    bf16* vW1T  = (bf16*)alloc((size_t)Hm * (2 * H) * 2);
    bf16* vW2T  = (bf16*)alloc((size_t)NV * Hm * 2);
    bf16* dW1T  = (bf16*)alloc((size_t)Hm * (2 * H) * 2);
    bf16* dW2T  = (bf16*)alloc((size_t)(d * d) * Hm * 2);
    float* hf   = (float*)alloc((size_t)4 * Bsz * H * 4);  // [dir][parity]
    bf16* hb    = (bf16*)alloc((size_t)4 * Bsz * H * 2);
    bf16* hcat  = (bf16*)alloc((size_t)BT * (2 * H) * 2);
    bf16* tm    = (bf16*)alloc((size_t)BT * Hm * 2);
    bf16* tv    = (bf16*)alloc((size_t)BT * Hm * 2);
    bf16* td    = (bf16*)alloc((size_t)BT * Hm * 2);
    float* vfl  = (float*)alloc((size_t)BT * NV * 4);
    unsigned* ctr = (unsigned*)alloc(2 * sizeof(unsigned));
    (void)ws_size; (void)in_sizes; (void)n_in; (void)out_size;

    auto cvt = [&](const float* src, bf16* dst, long n) {
        cvt_f32_to_bf16_kernel<<<(unsigned)((n + 255) / 256), 256, 0, stream>>>(src, dst, n);
    };
    auto tpose = [&](const float* W, bf16* Wt, int K, int N) {
        long n = (long)K * N;
        transpose_cvt_kernel<<<(unsigned)((n + 255) / 256), 256, 0, stream>>>(W, Wt, K, N);
    };
    auto gemm = [&](const bf16* A, const bf16* Bt, const float* bias,
                    float* Cf, bf16* Cb, int M, int N, int K, int act) {
        dim3 grid((N + 63) / 64, M / 32);
        gemm_bf16_kernel<<<grid, 256, 0, stream>>>(A, Bt, bias, Cf, Cb, M, N, K, act);
    };

    // ---- phase 0: precision conversion + weight transposes (one-time cost)
    cvt(x, xb, (long)BT * Din);
    tpose(iW1, iW1T, Din, Hm);
    tpose(iW2, iW2T, Hm, H);
    tpose(fWi, fWiT, H, H3);
    tpose(fWh, fWhT, H, H3);
    tpose(bWi, bWiT, H, H3);
    tpose(bWh, bWhT, H, H3);
    tpose(mW1, mW1T, 2 * H, Hm);
    tpose(mW2, mW2T, Hm, d);
    tpose(vW1, vW1T, 2 * H, Hm);
    tpose(vW2, vW2T, Hm, NV);
    tpose(dW1, dW1T, 2 * H, Hm);
    tpose(dW2, dW2T, Hm, d * d);

    // ---- phase 1: input MLP + GRU input projections (token-parallel WMMA)
    gemm(xb,  iW1T, ib1, nullptr, t1b, BT, Hm, Din, 1);   // relu
    gemm(t1b, iW2T, ib2, nullptr, ub,  BT, H,  Hm,  0);   // u (bf16)
    gemm(ub,  fWiT, fbi, xpf, nullptr, BT, H3, H,   0);   // xp forward (f32)
    gemm(ub,  bWiT, bbi, xpb, nullptr, BT, H3, H,   0);   // xp backward (f32)

    // ---- phase 2: persistent bidirectional GRU recurrence
    zero_ctr_kernel<<<1, 32, 0, stream>>>(ctr);
    hipFuncSetAttribute((const void*)gru_kernel,
                        hipFuncAttributeMaxDynamicSharedMemorySize, GRU_SMEM);
    gru_kernel<<<2 * GRU_S, 256, GRU_SMEM, stream>>>(
        xpf, xpb, fbhn, bbhn, fWhT, bWhT, hf, hb, hcat, ctr);

    // ---- phase 3: head MLPs (token-parallel WMMA)
    gemm(hcat, mW1T, mb1, nullptr, tm, BT, Hm, 2 * H, 1);
    gemm(tm,   mW2T, mb2, outB, nullptr, BT, d, Hm, 0);        // b
    gemm(hcat, vW1T, vb1, nullptr, tv, BT, Hm, 2 * H, 1);
    gemm(tv,   vW2T, vb2, vfl, nullptr, BT, NV, Hm, 0);        // vflat
    gemm(hcat, dW1T, db1, nullptr, td, BT, Hm, 2 * H, 1);
    gemm(td,   dW2T, db2, outA, nullptr, BT, d * d, Hm, 0);    // A

    // ---- phase 4: Q = L L^T per token
    qbuild_kernel<<<BT / 8, 256, 0, stream>>>(vfl, outQ);
}
